// Block_51900384804924
// MI455X (gfx1250) — matrix operand order repaired and checked by reading, not compiled, not run on hardware
//
#include <hip/hip_runtime.h>

// ---------------------------------------------------------------------------
// Transformer block for MI455X (gfx1250, wave32, WMMA).
// f32 inputs -> bf16 WMMA (f32 accumulate) for all matmuls.
// GEMM K-tiles move HBM->LDS via the Tensor Data Mover (TENSORcnt),
// double-buffered so DMA of tile t+1 overlaps WMMA of tile t.
// Attention Q/K tiles use per-lane global_load_async_to_lds_b128 (ASYNCcnt).
// ---------------------------------------------------------------------------

typedef unsigned short u16;
typedef __attribute__((ext_vector_type(16))) __bf16 v16bf;
typedef __attribute__((ext_vector_type(8)))  float  v8f;
typedef __attribute__((ext_vector_type(4)))  unsigned int v4u;
typedef __attribute__((ext_vector_type(8)))  int    v8i;
typedef __attribute__((ext_vector_type(4)))  int    v4i;

union V8U { v8f v; float f[8]; };

#define N_EMB   1024
#define SEQ_T   2048
#define BATCH   4
#define D_FF    4096
#define M_ROWS  (BATCH * SEQ_T)          // 8192
#define ATT_SCALE 0.03125f               // 1024^-0.5 (full embed dim, per ref)

__device__ __forceinline__ u16 f2bf(float f) {
  unsigned u = __float_as_uint(f);
  u += 0x7fffu + ((u >> 16) & 1u);       // round-to-nearest-even
  return (u16)(u >> 16);
}

__device__ __forceinline__ v8f vzero8() {
  v8f z = {0.f, 0.f, 0.f, 0.f, 0.f, 0.f, 0.f, 0.f};
  return z;
}

__device__ __forceinline__ v8f wmma_bf16(v16bf a, v16bf b, v8f c) {
  // D = A(16x32 bf16) * B(32x16 bf16) + C(16x16 f32)
  return __builtin_amdgcn_wmma_f32_16x16x32_bf16(
      /*neg_a=*/false, a, /*neg_b=*/false, b,
      /*c_mod=*/(short)0, c, /*reuse_a=*/false, /*reuse_b=*/false);
}

// LDS aperture keeps the byte offset in addr[31:0] -> truncation is the offset
__device__ __forceinline__ unsigned lds_addr32(const void* p) {
  return (unsigned)(size_t)p;
}

// ---------------------------------------------------------------------------
// TDM: 2D tile load, tile_k x rows of 2-byte elements, row stride K elements.
// LDS padding: 4 DWORDs (16B) after every 16 DWORDs (64B = one 32-elem row)
// -> LDS row stride 80B, matching LDT=40 u16.
// D# group0: [1:0]=count=1, [63:32]=lds_addr, [120:64]=global_addr,
//            [127:126]=type=2.
// D# group1: [17:16]=data_size(2B)=1, [20]=pad_enable, [24:22]=pad_interval=3
//            (16 DW), [31:25]=pad_amount=3 (4 DW), tensor_dim0=K,
//            tensor_dim1=rows, tile_dim0=tile_k, tile_dim1=rows,
//            tensor_dim0_stride=K.
// ---------------------------------------------------------------------------
__device__ __forceinline__ void tdm_load_2d(unsigned lds_byte_addr,
                                            const void* gaddr,
                                            int K, int rows, int tile_k) {
  unsigned long long ga = (unsigned long long)(size_t)gaddr;
  v4u g0 = { 1u,
             lds_byte_addr,
             (unsigned)(ga & 0xFFFFFFFFull),
             (unsigned)((ga >> 32) & 0x1FFFFFFull) | 0x80000000u };
  v8i g1 = { (int)0x06D10000,              // data_size=2B, pad 16DW/4DW
             (int)((K & 0xFFFF) << 16),    // tensor_dim0[15:0]
             (int)(rows << 16),            // tensor_dim0[31:16]=0, dim1[15:0]
             (int)(tile_k << 16),          // dim1[31:16]=0, tile_dim0
             rows,                         // tile_dim1 (tile_dim2=0)
             K,                            // tensor_dim0_stride[31:0]
             0, 0 };                       // stride hi, dim1_stride
  v4i gz = { 0, 0, 0, 0 };
#if __clang_major__ >= 23
  v8i gz8 = { 0, 0, 0, 0, 0, 0, 0, 0 };
  __builtin_amdgcn_tensor_load_to_lds(g0, g1, gz, gz, gz8, 0);
#else
  __builtin_amdgcn_tensor_load_to_lds(g0, g1, gz, gz, 0);
#endif
}

// A fragment (16x32, 16-bit): lane = 16*half + (M%16).
// half 0: elems 0..15 -> K=0..15
// half 1: elems 0..15 -> K=16..31
__device__ __forceinline__ v16bf load_a_frag(const u16* lds, int row0, int koff,
                                             int stride, int lane) {
  int r  = row0 + (lane & 15);
  int hf = lane >> 4;
  union { uint4 q[2]; v16bf v; } u;
  u.q[0] = *(const uint4*)(lds + r * stride + koff + hf * 16);
  u.q[1] = *(const uint4*)(lds + r * stride + koff + hf * 16 + 8);
  return u.v;
}

// B fragment (32x16, 16-bit) from Bt (row = output col N, contiguous K):
// lanes 0-15 hold K=0..15 of col N=lane; lanes 16-31 hold K=16..31.
__device__ __forceinline__ v16bf load_b_frag(const u16* lds, int row0, int koff,
                                             int stride, int lane) {
  int r  = row0 + (lane & 15);
  int hf = lane >> 4;
  const uint4* p = (const uint4*)(lds + r * stride + koff + hf * 16);
  union { uint4 q[2]; v16bf v; } u;
  u.q[0] = p[0];
  u.q[1] = p[1];
  return u.v;
}

// ---------------------------------------------------------------------------
// Weight convert + transpose: W[K][N] f32 -> Wt[N][K] bf16
// ---------------------------------------------------------------------------
__global__ __launch_bounds__(256) void cvt_t_kernel(const float* __restrict__ W,
                                                    u16* __restrict__ Wt,
                                                    int K, int N) {
  __shared__ float tile[32][33];
  int tx = threadIdx.x & 31;
  int ty = threadIdx.x >> 5;              // 0..7
  int n0 = blockIdx.x * 32, k0 = blockIdx.y * 32;
#pragma unroll
  for (int j = 0; j < 4; j++)
    tile[ty + j * 8][tx] = W[(size_t)(k0 + ty + j * 8) * N + n0 + tx];
  __syncthreads();
#pragma unroll
  for (int j = 0; j < 4; j++)
    Wt[(size_t)(n0 + ty + j * 8) * K + k0 + tx] = f2bf(tile[tx][ty + j * 8]);
}

// ---------------------------------------------------------------------------
// LayerNorm: one row (1024 cols) per block of 256 threads; bf16 output
// ---------------------------------------------------------------------------
__global__ __launch_bounds__(256) void ln_kernel(const float* __restrict__ x,
                                                 const float* __restrict__ g,
                                                 const float* __restrict__ beta,
                                                 u16* __restrict__ out) {
  int row = blockIdx.x;
  int tid = threadIdx.x;
  const float4 xv = ((const float4*)(x + (size_t)row * N_EMB))[tid];
  float s  = xv.x + xv.y + xv.z + xv.w;
  float s2 = xv.x * xv.x + xv.y * xv.y + xv.z * xv.z + xv.w * xv.w;
#pragma unroll
  for (int off = 1; off < 32; off <<= 1) {
    s  += __shfl_xor(s, off, 32);
    s2 += __shfl_xor(s2, off, 32);
  }
  __shared__ float sred[8], sred2[8];
  if ((tid & 31) == 0) { sred[tid >> 5] = s; sred2[tid >> 5] = s2; }
  __syncthreads();
  float ts = 0.f, ts2 = 0.f;
#pragma unroll
  for (int i = 0; i < 8; i++) { ts += sred[i]; ts2 += sred2[i]; }
  float mu   = ts * (1.f / N_EMB);
  float var  = ts2 * (1.f / N_EMB) - mu * mu;
  float rinv = rsqrtf(var + 1e-5f);
  const float4 gv = ((const float4*)g)[tid];
  const float4 bv = ((const float4*)beta)[tid];
  ushort4 r;
  r.x = f2bf((xv.x - mu) * rinv * gv.x + bv.x);
  r.y = f2bf((xv.y - mu) * rinv * gv.y + bv.y);
  r.z = f2bf((xv.z - mu) * rinv * gv.z + bv.z);
  r.w = f2bf((xv.w - mu) * rinv * gv.w + bv.w);
  ((ushort4*)(out + (size_t)row * N_EMB))[tid] = r;
}

// ---------------------------------------------------------------------------
// Tiled WMMA GEMM: out = act( A[M][K]bf16 * Bt[N][K]bf16^T + bias ) (+ res)
// 256 thr (8 waves), 128x128 tile. K staged 32-wide via TDM to LDS,
// double-buffered: tile t+1 DMA overlaps tile t WMMA. Wave 0 drives the TDM.
// ---------------------------------------------------------------------------
#define LDT 40   // LDS row stride in u16 (32 + 8 pad), 80B (16B aligned)

__global__ __launch_bounds__(256) void gemm_kernel(
    const u16* __restrict__ A, const u16* __restrict__ Bt,
    const float* __restrict__ bias, const float* __restrict__ res,
    float* __restrict__ outf, u16* __restrict__ outb,
    int N, int K, int relu) {
  __shared__ u16 a_lds[2][128 * LDT];
  __shared__ u16 b_lds[2][128 * LDT];
  int tid  = threadIdx.x;
  int lane = tid & 31;
  int w    = tid >> 5;
  int wm   = w >> 1;                      // 0..3 (M)
  int wn   = w & 1;                       // 0..1 (N)
  int m0   = blockIdx.y * 128;
  int n0   = blockIdx.x * 128;

  v8f acc[2][4];
#pragma unroll
  for (int mi = 0; mi < 2; mi++)
#pragma unroll
    for (int ni = 0; ni < 4; ni++) acc[mi][ni] = vzero8();

  const u16* Abase = A  + (size_t)m0 * K;
  const u16* Bbase = Bt + (size_t)n0 * K;
  unsigned lds_a[2] = { lds_addr32(&a_lds[0][0]), lds_addr32(&a_lds[1][0]) };
  unsigned lds_b[2] = { lds_addr32(&b_lds[0][0]), lds_addr32(&b_lds[1][0]) };

  auto issue_tile = [&](int buf, int k0) {
    if (w == 0) {
      tdm_load_2d(lds_a[buf], Abase + k0, K, 128, 32);
      tdm_load_2d(lds_b[buf], Bbase + k0, K, 128, 32);
    }
  };

  issue_tile(0, 0);
  for (int k0 = 0; k0 < K; k0 += 32) {
    int cur = (k0 >> 5) & 1;
    __builtin_amdgcn_s_wait_tensorcnt(0); // wave 0: current tile landed
    __syncthreads();                      // collective: DMA visible, prev buf free
    if (k0 + 32 < K) issue_tile(cur ^ 1, k0 + 32);
    if (w == 0 && k0 + 64 < K) {          // warm L2 for tile t+2
      __builtin_prefetch((const void*)(Abase + k0 + 64), 0, 1);
      __builtin_prefetch((const void*)(Bbase + k0 + 64), 0, 1);
    }
    const u16* ab = &a_lds[cur][0];
    const u16* bb = &b_lds[cur][0];
    v16bf af0 = load_a_frag(ab, wm * 32,      0, LDT, lane);
    v16bf af1 = load_a_frag(ab, wm * 32 + 16, 0, LDT, lane);
#pragma unroll
    for (int ni = 0; ni < 4; ni++) {
      v16bf bfr = load_b_frag(bb, wn * 64 + ni * 16, 0, LDT, lane);
      acc[0][ni] = wmma_bf16(af0, bfr, acc[0][ni]);
      acc[1][ni] = wmma_bf16(af1, bfr, acc[1][ni]);
    }
  }

  // epilogue: C layout -> lane holds col n0+..+lane%16, rows v + 8*(lane/16)
  int lanelo = lane & 15, lhf = lane >> 4;
#pragma unroll
  for (int mi = 0; mi < 2; mi++) {
#pragma unroll
    for (int ni = 0; ni < 4; ni++) {
      V8U u; u.v = acc[mi][ni];
      int col = n0 + wn * 64 + ni * 16 + lanelo;
      float bval = bias ? bias[col] : 0.f;
#pragma unroll
      for (int v = 0; v < 8; v++) {
        int row = m0 + wm * 32 + mi * 16 + v + 8 * lhf;
        float val = u.f[v] + bval;
        if (relu) val = fmaxf(val, 0.f);
        size_t idx = (size_t)row * N + col;
        if (res) val += res[idx];
        if (outb) outb[idx] = f2bf(val);
        else      outf[idx] = val;
      }
    }
  }
}

// ---------------------------------------------------------------------------
// Flash-style causal attention: grid (T/64, B*H), 128 thr (4 waves).
// Wave w owns 16 query rows; S=Q*K^T and O+=P*V via WMMA; online softmax.
// Q and K tiles DMA'd to LDS via per-lane async loads; V transposed in VGPRs.
// ---------------------------------------------------------------------------
#define ALD 72   // LDS row stride in u16 (64 + 8 pad), 144B (16B aligned)

__global__ __launch_bounds__(128) void attn_kernel(const u16* __restrict__ qb,
                                                   const u16* __restrict__ kb,
                                                   const u16* __restrict__ vb,
                                                   u16* __restrict__ yb) {
  __shared__ u16 q_lds[64 * ALD];
  __shared__ u16 k_lds[64 * ALD];
  __shared__ u16 vt_lds[64 * ALD];        // V^T : [d][key]
  __shared__ u16 p_lds[64 * ALD];         // P   : [qrow][key]
  int tid = threadIdx.x;
  int lane = tid & 31, w = tid >> 5;
  int lanelo = lane & 15, lhf = lane >> 4;
  int qt = blockIdx.x;
  int b  = blockIdx.y >> 4;
  int h0 = (blockIdx.y & 15) * 64;
  int q0 = qt * 64;
  size_t base = (size_t)b * SEQ_T * N_EMB;

  int sr = tid >> 1, shf = tid & 1;       // staging: 2 threads/row, 64B each
  { // stage Q tile (64 x 64) via async DMA; waited at first kt iteration
    unsigned lq = lds_addr32(q_lds + sr * ALD + shf * 32);
    unsigned long long pq =
        (unsigned long long)(size_t)(qb + base + (size_t)(q0 + sr) * N_EMB + h0 + shf * 32);
    asm volatile(
        "global_load_async_to_lds_b128 %0, %1, off\n\t"
        "global_load_async_to_lds_b128 %0, %1, off offset:16\n\t"
        "global_load_async_to_lds_b128 %0, %1, off offset:32\n\t"
        "global_load_async_to_lds_b128 %0, %1, off offset:48"
        :: "v"(lq), "v"(pq) : "memory");
  }

  float m_[8], l_[8];
  v8f o[4];
#pragma unroll
  for (int v = 0; v < 8; v++) { m_[v] = -1e30f; l_[v] = 0.f; }
#pragma unroll
  for (int f = 0; f < 4; f++) o[f] = vzero8();

  unsigned lk = lds_addr32(k_lds + sr * ALD + shf * 32);

  for (int kt = 0; kt <= qt; kt++) {
    __syncthreads();                      // previous K/V tiles consumed
    { // K tile rows (already Bt layout) via async DMA
      unsigned long long pk =
          (unsigned long long)(size_t)(kb + base + (size_t)(kt * 64 + sr) * N_EMB + h0 + shf * 32);
      asm volatile(
          "global_load_async_to_lds_b128 %0, %1, off\n\t"
          "global_load_async_to_lds_b128 %0, %1, off offset:16\n\t"
          "global_load_async_to_lds_b128 %0, %1, off offset:32\n\t"
          "global_load_async_to_lds_b128 %0, %1, off offset:48"
          :: "v"(lk), "v"(pk) : "memory");
      // V transposed through VGPRs (per-lane DMA cannot transpose)
      const uint4* vs =
          (const uint4*)(vb + base + (size_t)(kt * 64 + sr) * N_EMB + h0 + shf * 32);
      union { uint4 q[4]; u16 e[32]; } vv;
      vv.q[0] = vs[0]; vv.q[1] = vs[1]; vv.q[2] = vs[2]; vv.q[3] = vs[3];
#pragma unroll
      for (int j = 0; j < 32; j++) vt_lds[(shf * 32 + j) * ALD + sr] = vv.e[j];
      asm volatile("s_wait_asynccnt 0x0" ::: "memory");   // Q (first iter) + K landed
    }
    __syncthreads();

    // S(16x64) = Q(16x64) * K^T, two 32-wide K chunks over d
    v16bf aq0 = load_a_frag(q_lds, w * 16, 0,  ALD, lane);
    v16bf aq1 = load_a_frag(q_lds, w * 16, 32, ALD, lane);
    V8U s[4];
#pragma unroll
    for (int f = 0; f < 4; f++) {
      v8f sv = vzero8();
      sv = wmma_bf16(aq0, load_b_frag(k_lds, f * 16, 0,  ALD, lane), sv);
      sv = wmma_bf16(aq1, load_b_frag(k_lds, f * 16, 32, ALD, lane), sv);
      s[f].v = sv;
    }
    // scale + causal mask
#pragma unroll
    for (int f = 0; f < 4; f++) {
      int kc = kt * 64 + f * 16 + lanelo;
#pragma unroll
      for (int v = 0; v < 8; v++) {
        float val = s[f].f[v] * ATT_SCALE;
        int qr = q0 + w * 16 + v + 8 * lhf;
        if (kc > qr) val = -1e30f;
        s[f].f[v] = val;
      }
    }
    // online softmax (row = (v, half); values across 16 lanes of the half)
    float corr[8];
#pragma unroll
    for (int v = 0; v < 8; v++) {
      float mx = fmaxf(fmaxf(s[0].f[v], s[1].f[v]), fmaxf(s[2].f[v], s[3].f[v]));
#pragma unroll
      for (int off = 1; off < 16; off <<= 1) mx = fmaxf(mx, __shfl_xor(mx, off, 32));
      float mnew = fmaxf(m_[v], mx);
      corr[v] = __expf(m_[v] - mnew);
      float rs = 0.f;
#pragma unroll
      for (int f = 0; f < 4; f++) {
        float e = __expf(s[f].f[v] - mnew);
        s[f].f[v] = e;
        rs += e;
      }
#pragma unroll
      for (int off = 1; off < 16; off <<= 1) rs += __shfl_xor(rs, off, 32);
      l_[v] = l_[v] * corr[v] + rs;
      m_[v] = mnew;
    }
#pragma unroll
    for (int f = 0; f < 4; f++) {
      V8U ou; ou.v = o[f];
#pragma unroll
      for (int v = 0; v < 8; v++) ou.f[v] *= corr[v];
      o[f] = ou.v;
    }
    // P: C layout -> A layout via this wave's private LDS region (in-order DS)
#pragma unroll
    for (int f = 0; f < 4; f++)
#pragma unroll
      for (int v = 0; v < 8; v++)
        p_lds[(w * 16 + v + 8 * lhf) * ALD + f * 16 + lanelo] = f2bf(s[f].f[v]);

    // O += P(16x64) * V(64x64)  (B-frag from V^T rows = d columns)
    v16bf ap0 = load_a_frag(p_lds, w * 16, 0,  ALD, lane);
    v16bf ap1 = load_a_frag(p_lds, w * 16, 32, ALD, lane);
#pragma unroll
    for (int f = 0; f < 4; f++) {
      o[f] = wmma_bf16(ap0, load_b_frag(vt_lds, f * 16, 0,  ALD, lane), o[f]);
      o[f] = wmma_bf16(ap1, load_b_frag(vt_lds, f * 16, 32, ALD, lane), o[f]);
    }
  }

  // normalize and store y (bf16, head-sliced row-major)
#pragma unroll
  for (int f = 0; f < 4; f++) {
    V8U ou; ou.v = o[f];
#pragma unroll
    for (int v = 0; v < 8; v++) {
      int row = q0 + w * 16 + v + 8 * lhf;
      yb[base + (size_t)row * N_EMB + h0 + f * 16 + lanelo] = f2bf(ou.f[v] / l_[v]);
    }
  }
}

// ---------------------------------------------------------------------------
// Host orchestration
// ---------------------------------------------------------------------------
extern "C" void kernel_launch(void* const* d_in, const int* in_sizes, int n_in,
                              void* d_out, int out_size, void* d_ws, size_t ws_size,
                              hipStream_t stream) {
  (void)in_sizes; (void)n_in; (void)out_size; (void)ws_size;
  const float* x     = (const float*)d_in[0];
  const float* Wq    = (const float*)d_in[1];
  const float* Wk    = (const float*)d_in[2];
  const float* Wv    = (const float*)d_in[3];
  const float* Wp    = (const float*)d_in[4];
  const float* bp    = (const float*)d_in[5];
  const float* W1    = (const float*)d_in[6];
  const float* b1    = (const float*)d_in[7];
  const float* W2    = (const float*)d_in[8];
  const float* b2    = (const float*)d_in[9];
  const float* g1    = (const float*)d_in[10];
  const float* bt1   = (const float*)d_in[11];
  const float* g2    = (const float*)d_in[12];
  const float* bt2   = (const float*)d_in[13];
  float* out = (float*)d_out;

  char* ws = (char*)d_ws;
  const size_t MB = 1024ull * 1024ull;
  u16* WqT = (u16*)(ws + 0 * MB);         // 2 MB each
  u16* WkT = (u16*)(ws + 2 * MB);
  u16* WvT = (u16*)(ws + 4 * MB);
  u16* WpT = (u16*)(ws + 6 * MB);
  u16* W1T = (u16*)(ws + 8 * MB);         // 8 MB
  u16* W2T = (u16*)(ws + 16 * MB);        // 8 MB
  char* bufA = ws + 24 * MB;              // 80 MB region, aliased over time
  u16* h   = (u16*)(bufA);                // 16 MB
  u16* qbf = (u16*)(bufA + 16 * MB);
  u16* kbf = (u16*)(bufA + 32 * MB);
  u16* vbf = (u16*)(bufA + 48 * MB);
  u16* ybf = (u16*)(bufA + 64 * MB);
  u16* ff1 = (u16*)(bufA);                // aliases h/q/k/v (dead by then)
  u16* h2  = (u16*)(bufA + 64 * MB);      // aliases y (dead by then)
  float* x1 = (float*)(ws + 104 * MB);    // 32 MB f32

  dim3 blk256(256), blk128(128);

  // weights -> bf16 transposed
  cvt_t_kernel<<<dim3(32, 32), blk256, 0, stream>>>(Wq, WqT, N_EMB, N_EMB);
  cvt_t_kernel<<<dim3(32, 32), blk256, 0, stream>>>(Wk, WkT, N_EMB, N_EMB);
  cvt_t_kernel<<<dim3(32, 32), blk256, 0, stream>>>(Wv, WvT, N_EMB, N_EMB);
  cvt_t_kernel<<<dim3(32, 32), blk256, 0, stream>>>(Wp, WpT, N_EMB, N_EMB);
  cvt_t_kernel<<<dim3(128, 32), blk256, 0, stream>>>(W1, W1T, N_EMB, D_FF);
  cvt_t_kernel<<<dim3(32, 128), blk256, 0, stream>>>(W2, W2T, D_FF, N_EMB);

  // h = LN1(x)
  ln_kernel<<<M_ROWS, blk256, 0, stream>>>(x, g1, bt1, h);

  // q,k,v = h @ W{q,k,v}
  dim3 g_1024(N_EMB / 128, M_ROWS / 128);
  gemm_kernel<<<g_1024, blk256, 0, stream>>>(h, WqT, nullptr, nullptr, nullptr, qbf, N_EMB, N_EMB, 0);
  gemm_kernel<<<g_1024, blk256, 0, stream>>>(h, WkT, nullptr, nullptr, nullptr, kbf, N_EMB, N_EMB, 0);
  gemm_kernel<<<g_1024, blk256, 0, stream>>>(h, WvT, nullptr, nullptr, nullptr, vbf, N_EMB, N_EMB, 0);

  // y = causal attention
  attn_kernel<<<dim3(SEQ_T / 64, BATCH * 16), blk128, 0, stream>>>(qbf, kbf, vbf, ybf);

  // x1 = x + y @ Wp + bp
  gemm_kernel<<<g_1024, blk256, 0, stream>>>(ybf, WpT, bp, x, x1, nullptr, N_EMB, N_EMB, 0);

  // h2 = LN2(x1)
  ln_kernel<<<M_ROWS, blk256, 0, stream>>>(x1, g2, bt2, h2);

  // ff1 = relu(h2 @ W1 + b1)
  gemm_kernel<<<dim3(D_FF / 128, M_ROWS / 128), blk256, 0, stream>>>(
      h2, W1T, b1, nullptr, nullptr, ff1, D_FF, N_EMB, 1);

  // out = x1 + ff1 @ W2 + b2
  gemm_kernel<<<g_1024, blk256, 0, stream>>>(ff1, W2T, b2, x1, out, nullptr, N_EMB, D_FF, 0);
}
